// GNN_Embed_33526514712709
// MI455X (gfx1250) — matrix operand order repaired and checked by reading, not compiled, not run on hardware
//
#include <hip/hip_runtime.h>
#include <hip/hip_bf16.h>

typedef __attribute__((ext_vector_type(16))) __bf16 bf16x16;
typedef __attribute__((ext_vector_type(8)))  __bf16 bf16x8;
typedef __attribute__((ext_vector_type(4)))  __bf16 bf16x4;
typedef __attribute__((ext_vector_type(8)))  float  v8f;

// Types for the async global->LDS copy builtin (per hipcc diagnostic: the
// builtin takes "int __vector(4) __device__ *" style pointers).
typedef int i32x4v __attribute__((vector_size(16)));
typedef __attribute__((address_space(1))) i32x4v* as1_i32x4;
typedef __attribute__((address_space(3))) i32x4v* as3_i32x4;

#define TILES_PER_WAVE 8                     // 8 x 16 = 128 edges per wave
#define EDGES_PER_BLOCK (4 * TILES_PER_WAVE * 16)   // 512
#define PACK_BYTES 65536                     // 8kc * 8nt * 32lane * 16half * 2B
#define EDGE_DYN_SHMEM (PACK_BYTES + 4 * 16 * 264 * 2)  // pack + A staging

// ---------------------------------------------------------------------------
// Utility kernels
// ---------------------------------------------------------------------------
__global__ void zero_f32(float* __restrict__ p, long n) {
    long i = (long)blockIdx.x * blockDim.x + threadIdx.x;
    if (i < n) p[i] = 0.f;
}

// Pack a row-major fp32 weight matrix W[nout=128, kin] into bf16 WMMA
// B-fragments.  Consumer reads, per (kc, nt) fragment, 16 contiguous halfs at
// pack + ((kc*8+nt)*32 + lane)*16.  Layout (contiguous k order, as the A loads) of B 32x16:
//   lane l<16  : column n = nt*16+l,    rows k = kc*32 + 0..15
//   lane l>=16 : column n = nt*16+l-16, rows k = kc*32 + 16..31
__global__ void pack_b_kernel(const float* __restrict__ W, __bf16* __restrict__ pack,
                              int kin, int total) {
    int idx = blockIdx.x * 256 + threadIdx.x;
    if (idx >= total) return;
    int i    = idx & 15;
    int lane = (idx >> 4) & 31;
    int nt   = (idx >> 9) & 7;
    int kc   = idx >> 12;
    int hi   = lane >> 4;
    int k    = kc * 32 + i + hi * 16;
    int n    = nt * 16 + (lane & 15);
    pack[idx] = (__bf16)W[(long)n * kin + k];
}

// ---------------------------------------------------------------------------
// BatchNorm (training-mode, biased variance)
// ---------------------------------------------------------------------------
__global__ void bn_stats_kernel(const float* __restrict__ h, float* __restrict__ stats, int n) {
    const int col = threadIdx.x & 127;
    const int sub = threadIdx.x >> 7;
    const long r0 = (long)blockIdx.x * 64;
    float s = 0.f, s2 = 0.f;
    for (int i = sub; i < 64; i += 2) {
        long r = r0 + i;
        if (r < n) {
            float v = h[r * 128 + col];
            s += v; s2 += v * v;
        }
    }
    __shared__ float red[256];
    red[threadIdx.x] = s;
    __syncthreads();
    if (sub == 0) s += red[128 + col];
    __syncthreads();
    red[threadIdx.x] = s2;
    __syncthreads();
    if (sub == 0) {
        s2 += red[128 + col];
        atomicAdd(&stats[col], s);
        atomicAdd(&stats[128 + col], s2);
    }
}

__global__ void bn_apply_kernel(float* __restrict__ h, const float* __restrict__ stats,
                                const float* __restrict__ gamma, const float* __restrict__ beta,
                                int n) {
    long i = (long)blockIdx.x * 256 + threadIdx.x;
    if (i >= (long)n * 128) return;
    int col = (int)(i & 127);
    float inv_n = 1.f / (float)n;
    float mean  = stats[col] * inv_n;
    float var   = stats[128 + col] * inv_n - mean * mean;
    h[i] = (h[i] - mean) * rsqrtf(var + 1e-5f) * gamma[col] + beta[col];
}

// ---------------------------------------------------------------------------
// Edge kernel.  Per block: stage the 64KB bf16 weight pack into LDS once
// (async-to-LDS path when available), then each wave runs 8 x 16-edge tiles:
// full 16x128 edge-MLP via 64 WMMAs per tile with B-fragments served from
// LDS, sigmoid gate via cross-lane reduce, atomic scatter of gated messages.
// ---------------------------------------------------------------------------
__launch_bounds__(128)
__global__ void edge_wmma_kernel(const float* __restrict__ h,      // [N,128]
                                 const int*   __restrict__ eidx,   // [2,E]
                                 const float* __restrict__ eattr,  // [E]
                                 const __bf16* __restrict__ Apack, // packed linA (k<256)
                                 const float* __restrict__ law,    // [128,257] raw (attr col)
                                 const float* __restrict__ lab,    // [128]
                                 const float* __restrict__ lbw,    // [128]
                                 const float* __restrict__ lbb,    // [1]
                                 float* __restrict__ prop,         // [N,128]
                                 int E, int Etot)
{
    extern __shared__ __align__(64) char dynsmem[];
    __bf16* sPack = (__bf16*)dynsmem;                                   // 64KB
    __bf16 (*sX)[16][264] = (__bf16 (*)[16][264])(dynsmem + PACK_BYTES); // A staging
    __shared__ int   sSrc[4][16], sDst[4][16], sAct[4][16];
    __shared__ float sAttr[4][16], sW[4][16];

    const int wave = threadIdx.x >> 5;
    const int lane = threadIdx.x & 31;
    const int hi   = lane >> 4;
    const int l15  = lane & 15;

    // ---- Cooperative stage of the weight pack into LDS (once per block) ----
    {
        const char* g = (const char*)Apack;
        char*       l = dynsmem;
#if __has_builtin(__builtin_amdgcn_global_load_async_to_lds_b128)
        for (int i = threadIdx.x; i < PACK_BYTES / 16; i += 128)
            __builtin_amdgcn_global_load_async_to_lds_b128(
                (as1_i32x4)(g + (size_t)i * 16),
                (as3_i32x4)(l + (size_t)i * 16),
                0, 0);
#if __has_builtin(__builtin_amdgcn_s_wait_asynccnt)
        __builtin_amdgcn_s_wait_asynccnt(0);
#else
        asm volatile("s_wait_asynccnt 0" ::: "memory");
#endif
#else
        for (int i = threadIdx.x; i < PACK_BYTES / 16; i += 128)
            *(float4*)(l + (size_t)i * 16) = *(const float4*)(g + (size_t)i * 16);
#endif
    }
    __syncthreads();

    const long blockBase = (long)blockIdx.x * EDGES_PER_BLOCK;

    #pragma unroll 1
    for (int t = 0; t < TILES_PER_WAVE; ++t) {
        const long tileBase = blockBase + ((long)wave * TILES_PER_WAVE + t) * 16;
        if (tileBase >= (long)Etot) break;   // uniform per wave

        // Edge metadata: lanes 0..15 handle one edge each (self-loops appended).
        if (lane < 16) {
            long ge = tileBase + lane;
            int s = 0, dd = 0, act = 0; float at = 0.f;
            if (ge < (long)Etot) {
                act = 1;
                if (ge < (long)E) {
                    s  = eidx[ge];
                    dd = eidx[(long)E + ge];
                    at = eattr[ge];
                } else {
                    int nn = (int)(ge - E);
                    s = nn; dd = nn;           // self-loop, attr = 0
                }
            }
            sSrc[wave][lane] = s; sDst[wave][lane] = dd;
            sAct[wave][lane] = act; sAttr[wave][lane] = at;
        }

        // Gather + bf16 convert + stage rows (DS ops are in-order per wave).
        #pragma unroll 4
        for (int el = 0; el < 16; ++el) {
            int s  = sSrc[wave][el];
            int dd = sDst[wave][el];
            float4 xi = *(const float4*)(h + (long)dd * 128 + lane * 4); // x_i = h[dst]
            float4 xj = *(const float4*)(h + (long)s  * 128 + lane * 4); // x_j = h[src]
            __bf16* row = &sX[wave][el][0];
            bf16x4 a = { (__bf16)xi.x, (__bf16)xi.y, (__bf16)xi.z, (__bf16)xi.w };
            bf16x4 b = { (__bf16)xj.x, (__bf16)xj.y, (__bf16)xj.z, (__bf16)xj.w };
            *(bf16x4*)(row + lane * 4)       = a;
            *(bf16x4*)(row + 128 + lane * 4) = b;
        }

        // hk[16 x 128] = [x_i|x_j] @ linA_w[:, :256]^T  via 8x8 WMMA tiles.
        v8f acc[8];
        #pragma unroll
        for (int nt = 0; nt < 8; ++nt) acc[nt] = {};

        const __bf16* arow = &sX[wave][l15][0];
        #pragma unroll
        for (int kc = 0; kc < 8; ++kc) {
            // A-fragment (16x32): lane<16 holds K {0..15}, lane>=16 {16..31}
            union { bf16x16 v; bf16x8 h8[2]; } afr;
            int kb = kc * 32 + hi * 16;
            afr.h8[0] = *(const bf16x8*)(arow + kb);
            afr.h8[1] = *(const bf16x8*)(arow + kb + 8);
            #pragma unroll
            for (int nt = 0; nt < 8; ++nt) {
                bf16x16 bf = *(const bf16x16*)(sPack + (((kc * 8 + nt) * 32 + lane) << 4));
                acc[nt] = __builtin_amdgcn_wmma_f32_16x16x32_bf16(
                    false, afr.v, false, bf, (short)0, acc[nt], false, false);
            }
        }

        // Epilogue: + attr * linA_w[:,256] + bias, relu, dot with linB_w.
        float atr[8];
        #pragma unroll
        for (int j = 0; j < 8; ++j) atr[j] = sAttr[wave][j + hi * 8];
        float part[8];
        #pragma unroll
        for (int j = 0; j < 8; ++j) part[j] = 0.f;
        #pragma unroll
        for (int nt = 0; nt < 8; ++nt) {
            int n = nt * 16 + l15;
            float alast = law[n * 257 + 256];
            float ab    = lab[n];
            float bw    = lbw[n];
            #pragma unroll
            for (int j = 0; j < 8; ++j) {
                float v = acc[nt][j] + atr[j] * alast + ab;
                v = fmaxf(v, 0.f);
                part[j] += v * bw;
            }
        }
        // Reduce over the 16 lanes holding the same edge rows.
        #pragma unroll
        for (int m = 1; m <= 8; m <<= 1) {
            #pragma unroll
            for (int j = 0; j < 8; ++j)
                part[j] += __shfl_xor(part[j], m, 32);
        }
        if (l15 == 0) {
            float bb = lbb[0];
            #pragma unroll
            for (int j = 0; j < 8; ++j) {
                float z = part[j] + bb;
                sW[wave][j + hi * 8] = 1.f / (1.f + __expf(-z));   // sigmoid
            }
        }

        // Scatter: prop[dst] += w * x_j   (fp32 atomics, L2-resident target)
        #pragma unroll 2
        for (int el = 0; el < 16; ++el) {
            if (!sAct[wave][el]) continue;
            float we = sW[wave][el];
            int s  = sSrc[wave][el];
            int dd = sDst[wave][el];
            float4 xj = *(const float4*)(h + (long)s * 128 + lane * 4);
            float* p = prop + (long)dd * 128 + lane * 4;
            atomicAdd(p + 0, we * xj.x);
            atomicAdd(p + 1, we * xj.y);
            atomicAdd(p + 2, we * xj.z);
            atomicAdd(p + 3, we * xj.w);
        }
    }
}

// ---------------------------------------------------------------------------
// Node GEMM: out[r,:] = act([Ha[r] | Hb[r]] @ Wpack + bias).  16 rows/wave,
// kc_count k-chunks of 32 (8 for [h|prop], 4 for final h-only projection).
// May run in-place with out == Hb (each wave stages its rows before writing).
// ---------------------------------------------------------------------------
__launch_bounds__(128)
__global__ void node_wmma_kernel(const float* __restrict__ Ha,
                                 const float* Hb,                  // nullable; may alias out
                                 const __bf16* __restrict__ Wpack,
                                 const float* __restrict__ bias,
                                 float* out,
                                 int n_rows, int kc_count, int do_relu)
{
    __shared__ __align__(32) __bf16 sA[4][16][264];
    const int wave = threadIdx.x >> 5;
    const int lane = threadIdx.x & 31;
    const int hi   = lane >> 4;
    const int l15  = lane & 15;
    const long tileBase = ((long)blockIdx.x * 4 + wave) * 16;
    if (tileBase >= n_rows) return;

    __builtin_prefetch(Wpack, 0, 1);

    #pragma unroll 4
    for (int el = 0; el < 16; ++el) {
        long r  = tileBase + el;
        long rr = (r < n_rows) ? r : (n_rows - 1);
        float4 va = *(const float4*)(Ha + rr * 128 + lane * 4);
        __bf16* row = &sA[wave][el][0];
        bf16x4 pa = { (__bf16)va.x, (__bf16)va.y, (__bf16)va.z, (__bf16)va.w };
        *(bf16x4*)(row + lane * 4) = pa;
        if (Hb) {
            float4 vb = *(const float4*)(Hb + rr * 128 + lane * 4);
            bf16x4 pb = { (__bf16)vb.x, (__bf16)vb.y, (__bf16)vb.z, (__bf16)vb.w };
            *(bf16x4*)(row + 128 + lane * 4) = pb;
        }
    }

    v8f acc[8];
    #pragma unroll
    for (int nt = 0; nt < 8; ++nt) acc[nt] = {};

    const __bf16* arow = &sA[wave][l15][0];
    for (int kc = 0; kc < kc_count; ++kc) {
        union { bf16x16 v; bf16x8 h8[2]; } afr;
        int kb = kc * 32 + hi * 16;
        afr.h8[0] = *(const bf16x8*)(arow + kb);
        afr.h8[1] = *(const bf16x8*)(arow + kb + 8);
        #pragma unroll
        for (int nt = 0; nt < 8; ++nt) {
            bf16x16 bf = *(const bf16x16*)(Wpack + (((kc * 8 + nt) * 32 + lane) << 4));
            acc[nt] = __builtin_amdgcn_wmma_f32_16x16x32_bf16(
                false, afr.v, false, bf, (short)0, acc[nt], false, false);
        }
    }

    #pragma unroll
    for (int nt = 0; nt < 8; ++nt) {
        int n = nt * 16 + l15;
        float b = bias[n];
        #pragma unroll
        for (int j = 0; j < 8; ++j) {
            long r = tileBase + j + hi * 8;
            if (r < n_rows) {
                float v = acc[nt][j] + b;
                if (do_relu) v = fmaxf(v, 0.f);
                out[r * 128 + n] = v;
            }
        }
    }
}

// ---------------------------------------------------------------------------
// Launch
// ---------------------------------------------------------------------------
extern "C" void kernel_launch(void* const* d_in, const int* in_sizes, int n_in,
                              void* d_out, int out_size, void* d_ws, size_t ws_size,
                              hipStream_t stream) {
    const float* x      = (const float*)d_in[0];
    const int*   eidx   = (const int*)  d_in[1];
    const float* eattr  = (const float*)d_in[2];
    const float* lin_w  = (const float*)d_in[3];
    const float* lin_b  = (const float*)d_in[4];
    const float* linA_w = (const float*)d_in[5];
    const float* linA_b = (const float*)d_in[6];
    const float* linB_w = (const float*)d_in[7];
    const float* linB_b = (const float*)d_in[8];
    const float* bn_g   = (const float*)d_in[9];
    const float* bn_b   = (const float*)d_in[10];
    const float* fin_w  = (const float*)d_in[11];
    const float* fin_b  = (const float*)d_in[12];

    const int N = in_sizes[0] / 128;   // 50000
    const int E = in_sizes[2];         // 800000
    const int Etot = E + N;

    // Workspace carve-up (256B aligned sections).
    char* ws = (char*)d_ws;
    size_t off = 0;
    auto alloc = [&](size_t bytes) -> void* {
        void* p = ws + off;
        off += (bytes + 255) & ~(size_t)255;
        return p;
    };
    float*  bufA  = (float*)alloc((size_t)N * 128 * 4);
    float*  bufB  = (float*)alloc((size_t)N * 128 * 4);
    float*  stats = (float*)alloc(256 * 4);
    __bf16* packA[3];
    __bf16* packL[3];
    for (int l = 0; l < 3; ++l) packA[l] = (__bf16*)alloc(PACK_BYTES);
    for (int l = 0; l < 3; ++l) packL[l] = (__bf16*)alloc(PACK_BYTES);
    __bf16* packF = (__bf16*)alloc(PACK_BYTES / 2);

    // Pre-pack weights into WMMA B-fragment bf16 layout (deterministic, cheap).
    for (int l = 0; l < 3; ++l) {
        pack_b_kernel<<<128, 256, 0, stream>>>(linA_w + (size_t)l * 128 * 257, packA[l], 257, 32768);
        pack_b_kernel<<<128, 256, 0, stream>>>(lin_w  + (size_t)l * 128 * 256, packL[l], 256, 32768);
    }
    pack_b_kernel<<<64, 256, 0, stream>>>(fin_w, packF, 128, 16384);

    const int edgeBlocks = (Etot + EDGES_PER_BLOCK - 1) / EDGES_PER_BLOCK;
    const int nodeBlocks = (N + 63) / 64;
    const int elemBlocks = (int)(((long)N * 128 + 255) / 256);

    const float* hcur = x;                               // layer-0 input, never mutated
    float* bufs[2] = { bufA, bufB };
    for (int l = 0; l < 3; ++l) {
        if (l > 0) {
            zero_f32<<<1, 256, 0, stream>>>(stats, 256);
            bn_stats_kernel<<<(N + 63) / 64, 256, 0, stream>>>(hcur, stats, N);
            bn_apply_kernel<<<elemBlocks, 256, 0, stream>>>((float*)hcur, stats,
                bn_g + (l - 1) * 128, bn_b + (l - 1) * 128, N);
        }
        float* prop = bufs[l & 1];
        zero_f32<<<elemBlocks, 256, 0, stream>>>(prop, (long)N * 128);
        edge_wmma_kernel<<<edgeBlocks, 128, EDGE_DYN_SHMEM, stream>>>(hcur, eidx, eattr,
            packA[l], linA_w + (size_t)l * 128 * 257, linA_b + l * 128,
            linB_w + l * 128, linB_b + l, prop, E, Etot);
        // h_next = relu([h | prop] @ lin_w^T + b), written in-place over prop
        node_wmma_kernel<<<nodeBlocks, 128, 0, stream>>>(hcur, prop, packL[l],
            lin_b + l * 128, prop, N, 8, 1);
        hcur = prop;
    }
    // Final projection: out = h @ final_w^T + final_b  (no relu)
    node_wmma_kernel<<<nodeBlocks, 128, 0, stream>>>(hcur, nullptr, packF,
        fin_b, (float*)d_out, N, 4, 0);
}
